// Classifier_72816875536719
// MI455X (gfx1250) — matrix operand order repaired and checked by reading, not compiled, not run on hardware
//
#include <hip/hip_runtime.h>

#define NNODES 50000
#define NEDGES 800000
#define NREL   12
#define DIM    64
#define MCOLS  (NREL * DIM)   // 768

typedef __attribute__((ext_vector_type(16))) __bf16 v16bf;
typedef __attribute__((ext_vector_type(8)))  __bf16 v8bf;
typedef __attribute__((ext_vector_type(8)))  float  v8f;

__device__ inline v16bf cat16(v8bf a, v8bf b) {
  return __builtin_shufflevector(a, b, 0, 1, 2, 3, 4, 5, 6, 7,
                                       8, 9, 10, 11, 12, 13, 14, 15);
}

// ---------------------------------------------------------------------------
// GEMM tile: C(16x16) += A(16x64) * B(64x16) using pre-split bf16 hi/lo
// operands (fp32-class accuracy: ah*bh + ah*bl + al*bh).
//   A_hi/A_lo : [rows, 64] bf16 row-major   (node features)
//   Bt_hi/Bt_lo: [cols, 64] bf16 (TRANSPOSED weights: col-major over K)
// Fragments use contiguous k = k0 + 16*(lane/16) + s for both A and B (wave32); all loads contiguous b128.
// ---------------------------------------------------------------------------
__device__ inline v8f bf16_split_gemm_tile(
    const __bf16* __restrict__ A_hi, const __bf16* __restrict__ A_lo, int rowBase,
    const __bf16* __restrict__ Bt_hi, const __bf16* __restrict__ Bt_lo, int col0,
    int lane, v8f c)
{
  const int lidx = lane & 15;
  const int hi   = lane >> 4;
  const size_t arow = (size_t)(rowBase + lidx) * DIM;
  const size_t bcol = (size_t)(col0 + lidx) * DIM;
#pragma unroll
  for (int k0 = 0; k0 < DIM; k0 += 32) {
    const int ak = k0 + hi * 16;   // A half-lane K contiguous
    v16bf a_h = cat16(*(const v8bf*)(A_hi + arow + ak),
                      *(const v8bf*)(A_hi + arow + ak + 8));
    v16bf a_l = cat16(*(const v8bf*)(A_lo + arow + ak),
                      *(const v8bf*)(A_lo + arow + ak + 8));
    const int bk = k0 + hi * 16;   // B half-lane K contiguous
    v16bf b_h = *(const v16bf*)(Bt_hi + bcol + bk);
    v16bf b_l = *(const v16bf*)(Bt_lo + bcol + bk);
    c = __builtin_amdgcn_wmma_f32_16x16x32_bf16(false, a_l, false, b_h, (short)0, c, false, false);
    c = __builtin_amdgcn_wmma_f32_16x16x32_bf16(false, a_h, false, b_l, (short)0, c, false, false);
    c = __builtin_amdgcn_wmma_f32_16x16x32_bf16(false, a_h, false, b_h, (short)0, c, false, false);
  }
  return c;
}

__device__ inline void store_tile(float* __restrict__ out, size_t rowStride,
                                  int rowBase, int col0, int lane, v8f c)
{
  const int lidx = lane & 15;
  const int hi   = lane >> 4;
#pragma unroll
  for (int i = 0; i < 8; ++i)
    out[(size_t)(rowBase + i + hi * 8) * rowStride + col0 + lidx] = c[i];
}

__device__ inline void split_store(float v, __bf16* hi, __bf16* lo, int idx)
{
  __bf16 h = (__bf16)v;
  hi[idx] = h;
  lo[idx] = (__bf16)(v - (float)h);
}

// ---------------------------------------------------------------------------
// Weight conversion: W[m][k][c] fp32  ->  Wt_hi/lo[m][c][k] bf16 (transposed)
// ---------------------------------------------------------------------------
__global__ __launch_bounds__(256) void convert_w_kernel(
    const float* __restrict__ W, __bf16* __restrict__ t_hi,
    __bf16* __restrict__ t_lo, int nmats)
{
  int idx = blockIdx.x * 256 + threadIdx.x;
  if (idx >= nmats * DIM * DIM) return;
  int m = idx >> 12, rem = idx & 4095, k = rem >> 6, cc = rem & 63;
  split_store(W[idx], t_hi, t_lo, (m << 12) + (cc << 6) + k);
}

// h0 = relu(x * W_emb + b_emb), stored directly as bf16 hi/lo
__global__ __launch_bounds__(256) void embed_convert_kernel(
    const float* __restrict__ x, const float* __restrict__ W_emb,
    const float* __restrict__ b_emb, __bf16* __restrict__ h_hi,
    __bf16* __restrict__ h_lo)
{
  int idx = blockIdx.x * 256 + threadIdx.x;          // N*D exact
  int n = idx >> 6, d = idx & 63;
  float v = x[n] * W_emb[d] + b_emb[d];
  split_store(v > 0.f ? v : 0.f, h_hi, h_lo, idx);
}

// relu(h) -> bf16 hi/lo for the next layer's GEMMs
__global__ __launch_bounds__(256) void relu_convert_kernel(
    const float* __restrict__ h, __bf16* __restrict__ h_hi,
    __bf16* __restrict__ h_lo)
{
  int idx = blockIdx.x * 256 + threadIdx.x;          // N*D exact
  float v = h[idx];
  split_store(v > 0.f ? v : 0.f, h_hi, h_lo, idx);
}

// ---------------------------------------------------------------------------
// msgs[n, r*64+e] = h[n,:] @ Wrel[r][:,e]   ([N,64]x[64,768] GEMM via WMMA)
// 1 block = 16 nodes, 8 waves, each wave 6 of 48 column tiles.
// ---------------------------------------------------------------------------
__global__ __launch_bounds__(256) void msgs_gemm_kernel(
    const __bf16* __restrict__ h_hi, const __bf16* __restrict__ h_lo,
    const __bf16* __restrict__ wrelT_hi, const __bf16* __restrict__ wrelT_lo,
    float* __restrict__ msgs)
{
  const int lane = threadIdx.x & 31;
  const int wave = threadIdx.x >> 5;
  const int rowBase = blockIdx.x * 16;
  for (int t = 0; t < 6; ++t) {
    const int ct = wave * 6 + t;                     // 0..47
    const int r  = ct >> 2;
    const int e0 = (ct & 3) << 4;
    v8f c = {};
    c = bf16_split_gemm_tile(h_hi, h_lo, rowBase,
                             wrelT_hi + (size_t)r * DIM * DIM,
                             wrelT_lo + (size_t)r * DIM * DIM, e0, lane, c);
    store_tile(msgs, MCOLS, rowBase, r * DIM + e0, lane, c);
  }
}

// out = h @ Wself + b (+ h @ Wskip); scatter base for segment_sum.
__global__ __launch_bounds__(128) void node_update_kernel(
    const __bf16* __restrict__ h_hi, const __bf16* __restrict__ h_lo,
    const __bf16* __restrict__ wselfT_hi, const __bf16* __restrict__ wselfT_lo,
    const float* __restrict__ bias,
    const __bf16* __restrict__ wskipT_hi, const __bf16* __restrict__ wskipT_lo,
    float* __restrict__ out)
{
  const int lane = threadIdx.x & 31;
  const int wave = threadIdx.x >> 5;
  const int rowBase = blockIdx.x * 16;
  const int e0 = wave << 4;
  v8f c = {};
  c = bf16_split_gemm_tile(h_hi, h_lo, rowBase, wselfT_hi, wselfT_lo, e0, lane, c);
  if (wskipT_hi != nullptr)   // skip input == layer input in this model
    c = bf16_split_gemm_tile(h_hi, h_lo, rowBase, wskipT_hi, wskipT_lo, e0, lane, c);
  float bv = bias[e0 + (lane & 15)];
#pragma unroll
  for (int i = 0; i < 8; ++i) c[i] += bv;
  store_tile(out, DIM, rowBase, e0, lane, c);
}

// ---------------------------------------------------------------------------
// Segment-softmax state + edge passes
// ---------------------------------------------------------------------------
__device__ inline unsigned enc_f(float f) {
  unsigned u = __float_as_uint(f);
  return (u & 0x80000000u) ? ~u : (u | 0x80000000u);
}
__device__ inline float dec_f(unsigned v) {
  unsigned u = (v & 0x80000000u) ? (v & 0x7fffffffu) : ~v;
  return __uint_as_float(u);
}

__global__ __launch_bounds__(256) void seg_init_kernel(unsigned* __restrict__ segmax,
                                                       float* __restrict__ segsum)
{
  int i = blockIdx.x * 256 + threadIdx.x;
  if (i < NNODES) { segmax[i] = 0u; segsum[i] = 0.f; }
}

__global__ __launch_bounds__(256) void edge_pass1_kernel(
    const int* __restrict__ src, const int* __restrict__ dst,
    const int* __restrict__ etype, const float* __restrict__ msgs,
    const float* __restrict__ gw, const float* __restrict__ gb,
    const float* __restrict__ av, float* __restrict__ coef,
    float* __restrict__ score, unsigned* __restrict__ segmax)
{
  const int e    = blockIdx.x * 8 + (threadIdx.x >> 5);   // E exact
  const int lane = threadIdx.x & 31;
  const float2* m2 = (const float2*)(msgs + ((size_t)src[e] * NREL + etype[e]) * DIM);
  float2 mv = m2[lane];
  float g = mv.x * gw[2 * lane] + mv.y * gw[2 * lane + 1];
  float s = mv.x * av[2 * lane] + mv.y * av[2 * lane + 1];
#pragma unroll
  for (int off = 16; off; off >>= 1) {                    // wave32 reduction
    g += __shfl_xor(g, off, 32);
    s += __shfl_xor(s, off, 32);
  }
  if (lane == 0) {
    g = 1.f / (1.f + expf(-(g + gb[0])));
    s = s > 0.f ? s : 0.01f * s;                          // leaky_relu(0.01)
    coef[e]  = g;
    score[e] = s;
    atomicMax(&segmax[dst[e]], enc_f(s));
  }
}

__global__ __launch_bounds__(256) void edge_pass2_kernel(
    const int* __restrict__ dst, const float* __restrict__ score,
    const unsigned* __restrict__ segmax, float* __restrict__ coef,
    float* __restrict__ segsum)
{
  int e = blockIdx.x * 256 + threadIdx.x;                 // E exact
  int d = dst[e];
  float ex = expf(score[e] - dec_f(segmax[d]));
  atomicAdd(&segsum[d], ex);
  coef[e] *= ex;                                          // gate * exp
}

__global__ __launch_bounds__(256) void edge_pass3_kernel(
    const int* __restrict__ src, const int* __restrict__ dst,
    const int* __restrict__ etype, const float* __restrict__ msgs,
    const float* __restrict__ coef, const float* __restrict__ segsum,
    float* __restrict__ out)
{
  const int e    = blockIdx.x * 8 + (threadIdx.x >> 5);
  const int lane = threadIdx.x & 31;
  const int d = dst[e];
  float w = coef[e] / (segsum[d] + 1e-16f);
  const float2* m2 = (const float2*)(msgs + ((size_t)src[e] * NREL + etype[e]) * DIM);
  float2 mv = m2[lane];
  float* o = out + (size_t)d * DIM + 2 * lane;
  atomicAdd(o,     mv.x * w);
  atomicAdd(o + 1, mv.y * w);
}

// ---------------------------------------------------------------------------
// Host orchestration
// ---------------------------------------------------------------------------
struct WsLayout {
  __bf16 *hA_hi, *hA_lo, *hB_hi, *hB_lo;
  float  *h1, *h2, *msgs, *coef, *score, *segsum;
  unsigned* segmax;
  __bf16 *wrelT_hi, *wrelT_lo, *wselfT_hi, *wselfT_lo, *wskipT_hi, *wskipT_lo;
};

static void run_layer(const __bf16* hin_hi, const __bf16* hin_lo,
                      const float* Wrel, const float* Wself, const float* bias,
                      const float* gw, const float* gb, const float* av,
                      const float* Wskip,                 // null -> no skip
                      float* h_out,
                      __bf16* hout_hi, __bf16* hout_lo,   // null -> no relu/convert
                      const WsLayout& L,
                      const int* src, const int* dst, const int* etype,
                      hipStream_t stream)
{
  convert_w_kernel<<<(NREL * DIM * DIM) / 256, 256, 0, stream>>>(
      Wrel, L.wrelT_hi, L.wrelT_lo, NREL);
  convert_w_kernel<<<(DIM * DIM) / 256, 256, 0, stream>>>(
      Wself, L.wselfT_hi, L.wselfT_lo, 1);
  if (Wskip)
    convert_w_kernel<<<(DIM * DIM) / 256, 256, 0, stream>>>(
        Wskip, L.wskipT_hi, L.wskipT_lo, 1);

  msgs_gemm_kernel<<<NNODES / 16, 256, 0, stream>>>(
      hin_hi, hin_lo, L.wrelT_hi, L.wrelT_lo, L.msgs);
  seg_init_kernel<<<(NNODES + 255) / 256, 256, 0, stream>>>(L.segmax, L.segsum);
  node_update_kernel<<<NNODES / 16, 128, 0, stream>>>(
      hin_hi, hin_lo, L.wselfT_hi, L.wselfT_lo, bias,
      Wskip ? L.wskipT_hi : nullptr, Wskip ? L.wskipT_lo : nullptr, h_out);
  edge_pass1_kernel<<<NEDGES / 8, 256, 0, stream>>>(
      src, dst, etype, L.msgs, gw, gb, av, L.coef, L.score, L.segmax);
  edge_pass2_kernel<<<NEDGES / 256, 256, 0, stream>>>(
      dst, L.score, L.segmax, L.coef, L.segsum);
  edge_pass3_kernel<<<NEDGES / 8, 256, 0, stream>>>(
      src, dst, etype, L.msgs, L.coef, L.segsum, h_out);
  if (hout_hi)
    relu_convert_kernel<<<NNODES * DIM / 256, 256, 0, stream>>>(
        h_out, hout_hi, hout_lo);
}

extern "C" void kernel_launch(void* const* d_in, const int* in_sizes, int n_in,
                              void* d_out, int out_size, void* d_ws, size_t ws_size,
                              hipStream_t stream)
{
  (void)in_sizes; (void)n_in; (void)out_size; (void)ws_size;
  const float* x     = (const float*)d_in[0];
  const int*   src   = (const int*)d_in[1];
  const int*   dst   = (const int*)d_in[2];
  const int*   etype = (const int*)d_in[3];
  const float* W_emb = (const float*)d_in[4];
  const float* b_emb = (const float*)d_in[5];
  const float* Wrel[3]  = {(const float*)d_in[6],  (const float*)d_in[12], (const float*)d_in[18]};
  const float* Wself[3] = {(const float*)d_in[7],  (const float*)d_in[13], (const float*)d_in[19]};
  const float* bias[3]  = {(const float*)d_in[8],  (const float*)d_in[14], (const float*)d_in[20]};
  const float* gw[3]    = {(const float*)d_in[9],  (const float*)d_in[15], (const float*)d_in[21]};
  const float* gb[3]    = {(const float*)d_in[10], (const float*)d_in[16], (const float*)d_in[22]};
  const float* av[3]    = {(const float*)d_in[11], (const float*)d_in[17], (const float*)d_in[23]};
  const float* Wskip2 = (const float*)d_in[24];
  const float* Wskip3 = (const float*)d_in[25];

  char* ws = (char*)d_ws;
  WsLayout L;
  L.hA_hi  = (__bf16*)(ws);                        //  6.4 MB
  L.hA_lo  = (__bf16*)(ws +   6400000);            //  6.4 MB
  L.hB_hi  = (__bf16*)(ws +  12800000);            //  6.4 MB
  L.hB_lo  = (__bf16*)(ws +  19200000);            //  6.4 MB
  L.h1     = (float*) (ws +  25600000);            // 12.8 MB
  L.h2     = (float*) (ws +  38400000);            // 12.8 MB
  L.msgs   = (float*) (ws +  51200000);            // 153.6 MB (L2-resident)
  L.coef   = (float*) (ws + 204800000);            //  3.2 MB
  L.score  = (float*) (ws + 208000000);            //  3.2 MB
  L.segmax = (unsigned*)(ws + 211200000);          //  0.2 MB
  L.segsum = (float*) (ws + 211400000);            //  0.2 MB
  L.wrelT_hi  = (__bf16*)(ws + 211600000);         //  96 KB
  L.wrelT_lo  = (__bf16*)(ws + 211600000 + 131072);
  L.wselfT_hi = (__bf16*)(ws + 211600000 + 262144);
  L.wselfT_lo = (__bf16*)(ws + 211600000 + 270336);
  L.wskipT_hi = (__bf16*)(ws + 211600000 + 278528);
  L.wskipT_lo = (__bf16*)(ws + 211600000 + 286720);

  embed_convert_kernel<<<NNODES * DIM / 256, 256, 0, stream>>>(
      x, W_emb, b_emb, L.hA_hi, L.hA_lo);

  // layer 1: in=A, out fp32 h1, relu -> B
  run_layer(L.hA_hi, L.hA_lo, Wrel[0], Wself[0], bias[0], gw[0], gb[0], av[0],
            nullptr, L.h1, L.hB_hi, L.hB_lo, L, src, dst, etype, stream);
  // layer 2: in=B (skip=B), out fp32 h2, relu -> A
  run_layer(L.hB_hi, L.hB_lo, Wrel[1], Wself[1], bias[1], gw[1], gb[1], av[1],
            Wskip2, L.h2, L.hA_hi, L.hA_lo, L, src, dst, etype, stream);
  // layer 3: in=A (skip=A), out -> d_out, no activation
  run_layer(L.hA_hi, L.hA_lo, Wrel[2], Wself[2], bias[2], gw[2], gb[2], av[2],
            Wskip3, (float*)d_out, nullptr, nullptr, L, src, dst, etype, stream);
}
